// MiMoV2DecoderLayer_30133490548819
// MI455X (gfx1250) — matrix operand order repaired and checked by reading, not compiled, not run on hardware
//
#include <hip/hip_runtime.h>
#include <stdint.h>

// ---------------- constants ----------------
constexpr int T   = 2048;
constexpr int H   = 4096;
constexpr int NQ  = 32;
constexpr int NKV = 8;
constexpr int HD  = 128;
constexpr int VD  = 96;
constexpr int WIN = 1024;
constexpr int E   = 8;
constexpr int FF  = 1408;
constexpr int NQKV = NQ*HD + NKV*HD + NKV*VD;   // 5888
constexpr int CAP  = 2*T + 128*E;               // 5120 padded MoE slots

// ---------------- WMMA types ----------------
typedef __bf16 v16bf __attribute__((ext_vector_type(16)));
typedef float  v8f   __attribute__((ext_vector_type(8)));
union Frag16 { uint4 u[2]; v16bf v; };

__device__ __forceinline__ v8f vzero8() { v8f z = {0.f,0.f,0.f,0.f,0.f,0.f,0.f,0.f}; return z; }

__device__ __forceinline__ unsigned short f2bf(float f) {
  uint32_t u = __float_as_uint(f);
  uint32_t r = u + 0x7FFFu + ((u >> 16) & 1u);   // round-to-nearest-even
  return (unsigned short)(r >> 16);
}
__device__ __forceinline__ float bf2f(unsigned short h) {
  return __uint_as_float(((uint32_t)h) << 16);
}

// ---------------- workspace layout (bytes) ----------------
constexpr size_t SZ_XBF   = (size_t)T*H*2;
constexpr size_t SZ_QBF   = (size_t)NQ*T*HD*2;
constexpr size_t SZ_KBF   = (size_t)NKV*T*HD*2;
constexpr size_t SZ_VTBF  = (size_t)NKV*VD*T*2;
constexpr size_t SZ_ATTN  = (size_t)T*NQ*VD*2;
constexpr size_t SZ_WOT   = (size_t)H*(NQ*VD)*2;
constexpr size_t SZ_WQKVT = (size_t)NQKV*H*2;
constexpr size_t SZ_QKV   = (size_t)T*NQKV*4;
constexpr size_t SZ_H     = (size_t)T*H*4;
constexpr size_t SZ_X2BF  = (size_t)T*H*2;
constexpr size_t SZ_WGUP  = (size_t)E*2*FF*H*2;
constexpr size_t SZ_WDN   = (size_t)E*H*FF*2;

constexpr size_t O_XBF   = 0;
constexpr size_t O_QBF   = O_XBF  + SZ_XBF;
constexpr size_t O_KBF   = O_QBF  + SZ_QBF;
constexpr size_t O_VTBF  = O_KBF  + SZ_KBF;
constexpr size_t O_ATTN  = O_VTBF + SZ_VTBF;
constexpr size_t O_WOT   = O_ATTN + SZ_ATTN;
constexpr size_t O_WQKVT = O_WOT  + SZ_WOT;      // end of alias region 1
constexpr size_t O_QKV   = O_WQKVT + SZ_WQKVT;
constexpr size_t O_H     = O_QKV  + SZ_QKV;      // end of alias region 2
constexpr size_t O_X2BF  = O_H    + SZ_H;
constexpr size_t O_WGUP  = O_X2BF + SZ_X2BF;
constexpr size_t O_WDN   = O_WGUP + SZ_WGUP;
constexpr size_t O_SMALL = O_WDN  + SZ_WDN;

// aliases: h12/act reuse [x_bf..woT] (dead after attn+O-proj), eo reuses [wqkvT+qkv]
constexpr size_t O_H12 = O_XBF;
constexpr size_t SZ_H12 = (size_t)CAP*2*FF*4;
constexpr size_t O_ACT = O_H12 + SZ_H12;
constexpr size_t SZ_ACT = (size_t)CAP*FF*2;
constexpr size_t O_EO  = O_WQKVT;
constexpr size_t SZ_EO = (size_t)CAP*H*4;
static_assert(SZ_H12 <= O_WQKVT, "h12 alias overflow");
static_assert(O_ACT + SZ_ACT <= O_WQKVT, "act alias overflow");
static_assert(SZ_EO <= SZ_WQKVT + SZ_QKV, "eo alias overflow");

constexpr size_t O_IDS     = O_SMALL;
constexpr size_t O_WTS     = O_IDS     + (size_t)2*T*4;
constexpr size_t O_COUNTS  = O_WTS     + (size_t)2*T*4;
constexpr size_t O_CURSOR  = O_COUNTS  + 64;
constexpr size_t O_OFFS    = O_CURSOR  + 64;
constexpr size_t O_TOTAL   = O_OFFS    + 64;
constexpr size_t O_TOKSLOT = O_TOTAL   + 64;
constexpr size_t O_SLOTTOK = O_TOKSLOT + (size_t)CAP*4;

// ---------------- transpose + f32->bf16 convert ----------------
// in: R x C (row-major f32), out: C x R (row-major bf16 "Bt" layout)
__global__ void transpose_cvt_kernel(const float* __restrict__ in,
                                     unsigned short* __restrict__ out,
                                     int R, int C, long inStride, long outStride) {
  __shared__ float tile[32][33];
  const float* inp = in + (long)blockIdx.z * inStride;
  unsigned short* outp = out + (long)blockIdx.z * outStride;
  const int c0 = blockIdx.x * 32, r0 = blockIdx.y * 32;
  const int tx = threadIdx.x, ty = threadIdx.y;  // 32 x 8
#pragma unroll
  for (int i = 0; i < 4; ++i)
    tile[ty + 8*i][tx] = inp[(long)(r0 + ty + 8*i) * C + c0 + tx];
  __syncthreads();
#pragma unroll
  for (int i = 0; i < 4; ++i)
    outp[(long)(c0 + ty + 8*i) * R + r0 + tx] = f2bf(tile[tx][ty + 8*i]);
}

// ---------------- RMSNorm -> bf16 ----------------
__global__ void rmsnorm_kernel(const float* __restrict__ x, const float* __restrict__ w,
                               unsigned short* __restrict__ out) {
  const int t = blockIdx.x, tid = threadIdx.x;
  const float* row = x + (long)t * H;
  float s = 0.f;
  for (int i = tid; i < H; i += 256) { float v = row[i]; s += v * v; }
#pragma unroll
  for (int off = 16; off > 0; off >>= 1) s += __shfl_xor(s, off, 32);
  __shared__ float sm[8];
  if ((tid & 31) == 0) sm[tid >> 5] = s;
  __syncthreads();
  float tot = 0.f;
#pragma unroll
  for (int i = 0; i < 8; ++i) tot += sm[i];
  const float scale = rsqrtf(tot / (float)H + 1e-6f);
  for (int i = tid; i < H; i += 256) out[(long)t * H + i] = f2bf(row[i] * scale * w[i]);
}

// ---------------- generic bf16 WMMA GEMM ----------------
// C[M,N] = A[M,K] * Bt[N,K]^T ; 128x128 block tile, 8 waves (4x2), 32x64 per wave,
// K-step 32, double-buffered LDS, v_wmma_f32_16x16x32_bf16.
template <bool GROUPED, bool GATHER, bool ADD>
__global__ __launch_bounds__(256) void gemm_bf16_kernel(
    const unsigned short* __restrict__ A,
    const unsigned short* __restrict__ Bt,
    float* __restrict__ C, int ldc,
    const float* __restrict__ addend,
    int Kdim, long bExpertStride,
    const int* __restrict__ counts,
    const int* __restrict__ offsets,
    const int* __restrict__ tok_of_slot) {
  int base = 0;
  if (GROUPED) {
    const int e = blockIdx.z;
    const int cnt = counts[e];
    if ((int)blockIdx.y * 128 >= cnt) return;
    base = offsets[e];
    Bt += (long)e * bExpertStride;
  }
  const int tid = threadIdx.x;
  __shared__ __align__(16) unsigned short As[2][128][40];
  __shared__ __align__(16) unsigned short Bs[2][128][40];
  const int m0 = blockIdx.y * 128;
  const int n0 = blockIdx.x * 128;

  // global->LDS loader: 2 threads per row, 32B each
  const int lrow = tid >> 1;
  const int lseg = (tid & 1) * 16;  // element offset (bf16)
  long aRow;
  bool aValid = true;
  if (GATHER) {
    const int tok = tok_of_slot[base + m0 + lrow];
    aValid = tok >= 0;
    aRow = aValid ? (long)tok : 0;
  } else if (GROUPED) {
    aRow = (long)(base + m0 + lrow);
  } else {
    aRow = (long)(m0 + lrow);
  }
  const unsigned short* aPtr = A + aRow * (long)Kdim + lseg;
  const unsigned short* bPtr = Bt + (long)(n0 + lrow) * Kdim + lseg;

  const int lane = tid & 31;
  const int wid = tid >> 5;
  const int wm = (wid >> 1) * 32;  // 0,32,64,96
  const int wn = (wid & 1) * 64;   // 0,64
  const int lHalf = lane >> 4;
  const int lMod = lane & 15;

  v8f acc[2][4];
#pragma unroll
  for (int i = 0; i < 2; ++i)
#pragma unroll
    for (int j = 0; j < 4; ++j) acc[i][j] = vzero8();

  uint4 ra0, ra1, rb0, rb1;
  ra0 = *(const uint4*)(aPtr);
  ra1 = *(const uint4*)(aPtr + 8);
  rb0 = *(const uint4*)(bPtr);
  rb1 = *(const uint4*)(bPtr + 8);
  if (GATHER && !aValid) { ra0 = make_uint4(0,0,0,0); ra1 = ra0; }
  *(uint4*)&As[0][lrow][lseg]     = ra0;
  *(uint4*)&As[0][lrow][lseg + 8] = ra1;
  *(uint4*)&Bs[0][lrow][lseg]     = rb0;
  *(uint4*)&Bs[0][lrow][lseg + 8] = rb1;
  __syncthreads();

  const int kTiles = Kdim >> 5;
  for (int kt = 0; kt < kTiles; ++kt) {
    const int cur = kt & 1, nxt = cur ^ 1;
    if (kt + 1 < kTiles) {
      const unsigned short* ap = aPtr + (long)(kt + 1) * 32;
      const unsigned short* bp = bPtr + (long)(kt + 1) * 32;
      ra0 = *(const uint4*)(ap);
      ra1 = *(const uint4*)(ap + 8);
      rb0 = *(const uint4*)(bp);
      rb1 = *(const uint4*)(bp + 8);
      if (GATHER && !aValid) { ra0 = make_uint4(0,0,0,0); ra1 = ra0; }
    }
    if (kt + 2 < kTiles) {   // gfx1250 global_prefetch for tile after next
      __builtin_prefetch(aPtr + (long)(kt + 2) * 32, 0, 0);
      __builtin_prefetch(bPtr + (long)(kt + 2) * 32, 0, 0);
    }
    Frag16 af[2], bfr[4];
#pragma unroll
    for (int i = 0; i < 2; ++i) {
      const unsigned short* p = &As[cur][wm + i * 16 + lMod][0];
      af[i].u[0] = *(const uint4*)(p + lHalf * 8);
      af[i].u[1] = *(const uint4*)(p + 16 + lHalf * 8);
    }
#pragma unroll
    for (int j = 0; j < 4; ++j) {
      const unsigned short* p = &Bs[cur][wn + j * 16 + lMod][0];
      bfr[j].u[0] = *(const uint4*)(p + lHalf * 8);
      bfr[j].u[1] = *(const uint4*)(p + 16 + lHalf * 8);
    }
#pragma unroll
    for (int i = 0; i < 2; ++i)
#pragma unroll
      for (int j = 0; j < 4; ++j)
        acc[i][j] = __builtin_amdgcn_wmma_f32_16x16x32_bf16(
            false, af[i].v, false, bfr[j].v, (short)0, acc[i][j], false, false);
    __syncthreads();
    if (kt + 1 < kTiles) {
      *(uint4*)&As[nxt][lrow][lseg]     = ra0;
      *(uint4*)&As[nxt][lrow][lseg + 8] = ra1;
      *(uint4*)&Bs[nxt][lrow][lseg]     = rb0;
      *(uint4*)&Bs[nxt][lrow][lseg + 8] = rb1;
      __syncthreads();
    }
  }

  const long cRowBase = (long)(GROUPED ? (base + m0) : m0);
#pragma unroll
  for (int i = 0; i < 2; ++i) {
#pragma unroll
    for (int r = 0; r < 8; ++r) {
      const long row = cRowBase + wm + i * 16 + r + 8 * lHalf;
      const long col = (long)n0 + wn + lMod;
#pragma unroll
      for (int j = 0; j < 4; ++j) {
        float v = acc[i][j][r];
        const long idx = row * (long)ldc + col + j * 16;
        if (ADD) v += addend[idx];
        C[idx] = v;
      }
    }
  }
}

// ---------------- RoPE + repack q/k/vT to bf16 ----------------
__global__ void rope_pack_kernel(const float* __restrict__ qkv,
                                 unsigned short* __restrict__ qbf,
                                 unsigned short* __restrict__ kbf,
                                 unsigned short* __restrict__ vtbf) {
  const int t = blockIdx.x;
  const float* row = qkv + (long)t * NQKV;
  const float qscale = 0.08838834764831845f;           // HD^-0.5
  const float lnTheta_o64 = 13.815510557964274f / 64.f; // ln(1e6)/64
  for (int idx = threadIdx.x; idx < 2048 + 512 + 768; idx += 256) {
    if (idx < 2048) {                                   // Q: 32 heads x 64 pairs
      const int hh = idx >> 6, p = idx & 63;
      const float ang = (float)t * __expf(-(float)p * lnTheta_o64);
      float s, c; __sincosf(ang, &s, &c);
      const float x1 = row[hh * 128 + p], x2 = row[hh * 128 + 64 + p];
      const long base = ((long)hh * T + t) * HD;
      qbf[base + p]      = f2bf((x1 * c - x2 * s) * qscale);
      qbf[base + 64 + p] = f2bf((x2 * c + x1 * s) * qscale);
    } else if (idx < 2048 + 512) {                      // K: 8 heads x 64 pairs
      const int j = idx - 2048, hh = j >> 6, p = j & 63;
      const float ang = (float)t * __expf(-(float)p * lnTheta_o64);
      float s, c; __sincosf(ang, &s, &c);
      const float x1 = row[4096 + hh * 128 + p], x2 = row[4096 + hh * 128 + 64 + p];
      const long base = ((long)hh * T + t) * HD;
      kbf[base + p]      = f2bf(x1 * c - x2 * s);
      kbf[base + 64 + p] = f2bf(x2 * c + x1 * s);
    } else {                                            // V transposed: (h, vd, t)
      const int j = idx - 2560, hh = j / 96, d = j % 96;
      vtbf[((long)hh * VD + d) * T + t] = f2bf(row[5120 + hh * 96 + d]);
    }
  }
}

// ---------------- flash attention, 1 wave per (head, 16-row q tile) ----------------
__global__ __launch_bounds__(32) void attn_kernel(
    const unsigned short* __restrict__ qbf,   // [NQ][T][HD] (pre-scaled)
    const unsigned short* __restrict__ kbf,   // [NKV][T][HD]
    const unsigned short* __restrict__ vtbf,  // [NKV][VD][T]
    const float* __restrict__ sink,
    unsigned short* __restrict__ attn) {      // [T][NQ*VD] bf16
  const int h = blockIdx.y;
  const int q0 = blockIdx.x * 16;
  const int kvh = h >> 2;
  const int lane = threadIdx.x;
  const int lHalf = lane >> 4, lMod = lane & 15;

  __shared__ __align__(16) unsigned short P[16][40];

  Frag16 qf[4];
  const unsigned short* qrow = qbf + ((long)h * T + q0 + lMod) * HD;
#pragma unroll
  for (int c = 0; c < 4; ++c) {
    qf[c].u[0] = *(const uint4*)(qrow + c * 32 + lHalf * 8);
    qf[c].u[1] = *(const uint4*)(qrow + c * 32 + 16 + lHalf * 8);
  }
  float m_i[8], l_i[8];
  v8f acc[6];
  const float snk = sink[h];
#pragma unroll
  for (int r = 0; r < 8; ++r) { m_i[r] = snk; l_i[r] = 1.f; }  // sink column folded in
#pragma unroll
  for (int v = 0; v < 6; ++v) acc[v] = vzero8();

  int jlo = q0 - (WIN - 1); if (jlo < 0) jlo = 0; jlo &= ~31;
  for (int kt = jlo; kt <= q0 + 15; kt += 32) {
    v8f S0 = vzero8(), S1 = vzero8();
    const unsigned short* krow0 = kbf + ((long)kvh * T + kt + lMod) * HD;
    const unsigned short* krow1 = krow0 + 16 * HD;
#pragma unroll
    for (int c = 0; c < 4; ++c) {
      Frag16 kf0, kf1;
      kf0.u[0] = *(const uint4*)(krow0 + c * 32 + lHalf * 8);
      kf0.u[1] = *(const uint4*)(krow0 + c * 32 + 16 + lHalf * 8);
      kf1.u[0] = *(const uint4*)(krow1 + c * 32 + lHalf * 8);
      kf1.u[1] = *(const uint4*)(krow1 + c * 32 + 16 + lHalf * 8);
      S0 = __builtin_amdgcn_wmma_f32_16x16x32_bf16(false, qf[c].v, false, kf0.v, (short)0, S0, false, false);
      S1 = __builtin_amdgcn_wmma_f32_16x16x32_bf16(false, qf[c].v, false, kf1.v, (short)0, S1, false, false);
    }
    // causal + sliding-window mask using C layout (row = r + 8*lHalf, col = lMod)
    const int iBase = q0 + 8 * lHalf;
    const int j0c = kt + lMod, j1c = j0c + 16;
    float alpha[8];
#pragma unroll
    for (int r = 0; r < 8; ++r) {
      const int i = iBase + r;
      if (j0c > i || (i - j0c) >= WIN) S0[r] = -1e30f;
      if (j1c > i || (i - j1c) >= WIN) S1[r] = -1e30f;
      float mx = fmaxf(S0[r], S1[r]);
#pragma unroll
      for (int off = 1; off < 16; off <<= 1) mx = fmaxf(mx, __shfl_xor(mx, off, 16));
      const float mn = fmaxf(m_i[r], mx);
      alpha[r] = __expf(m_i[r] - mn);
      m_i[r] = mn;
      const float p0 = __expf(S0[r] - mn);
      const float p1 = __expf(S1[r] - mn);
      S0[r] = p0; S1[r] = p1;
      float rs = p0 + p1;
#pragma unroll
      for (int off = 1; off < 16; off <<= 1) rs += __shfl_xor(rs, off, 16);
      l_i[r] = l_i[r] * alpha[r] + rs;
    }
#pragma unroll
    for (int v = 0; v < 6; ++v)
#pragma unroll
      for (int r = 0; r < 8; ++r) acc[v][r] *= alpha[r];
    // P (C layout) -> LDS -> A fragment layout
#pragma unroll
    for (int r = 0; r < 8; ++r) {
      P[r + 8 * lHalf][lMod]      = f2bf(S0[r]);
      P[r + 8 * lHalf][lMod + 16] = f2bf(S1[r]);
    }
    __syncthreads();
    Frag16 pf;
    {
      const unsigned short* p = &P[lMod][0];
      pf.u[0] = *(const uint4*)(p + lHalf * 8);
      pf.u[1] = *(const uint4*)(p + 16 + lHalf * 8);
    }
#pragma unroll
    for (int v = 0; v < 6; ++v) {
      Frag16 vf;
      const unsigned short* vr = vtbf + ((long)kvh * VD + v * 16 + lMod) * T + kt;
      vf.u[0] = *(const uint4*)(vr + lHalf * 8);
      vf.u[1] = *(const uint4*)(vr + 16 + lHalf * 8);
      acc[v] = __builtin_amdgcn_wmma_f32_16x16x32_bf16(false, pf.v, false, vf.v, (short)0, acc[v], false, false);
    }
    __syncthreads();
  }
#pragma unroll
  for (int r = 0; r < 8; ++r) {
    const float inv = 1.f / l_i[r];
    const int row = q0 + r + 8 * lHalf;
#pragma unroll
    for (int v = 0; v < 6; ++v)
      attn[(long)row * (NQ * VD) + h * VD + v * 16 + lMod] = f2bf(acc[v][r] * inv);
  }
}

// ---------------- MoE routing ----------------
__global__ void router_kernel(const unsigned short* __restrict__ x2,
                              const float* __restrict__ gw,
                              int* __restrict__ ids, float* __restrict__ wts,
                              int* __restrict__ counts) {
  const int t = blockIdx.x, tid = threadIdx.x;
  float acc[8] = {0, 0, 0, 0, 0, 0, 0, 0};
  for (int i = tid; i < H; i += 256) {
    const float xv = bf2f(x2[(long)t * H + i]);
#pragma unroll
    for (int e = 0; e < 8; ++e) acc[e] += xv * gw[i * 8 + e];
  }
  __shared__ float sm[8][8];
#pragma unroll
  for (int e = 0; e < 8; ++e) {
    float v = acc[e];
#pragma unroll
    for (int off = 1; off < 32; off <<= 1) v += __shfl_xor(v, off, 32);
    if ((tid & 31) == 0) sm[tid >> 5][e] = v;
  }
  __syncthreads();
  if (tid == 0) {
    float lg[8];
#pragma unroll
    for (int e = 0; e < 8; ++e) {
      float v = 0.f;
#pragma unroll
      for (int w = 0; w < 8; ++w) v += sm[w][e];
      lg[e] = v;
    }
    int b0 = 0;
    for (int e = 1; e < 8; ++e) if (lg[e] > lg[b0]) b0 = e;
    int b1 = -1;
    for (int e = 0; e < 8; ++e) {
      if (e == b0) continue;
      if (b1 < 0 || lg[e] > lg[b1]) b1 = e;
    }
    // renormalized top-2 softmax == pairwise sigmoid of logits
    const float w0 = 1.f / (1.f + __expf(lg[b1] - lg[b0]));
    ids[t * 2] = b0; ids[t * 2 + 1] = b1;
    wts[t * 2] = w0; wts[t * 2 + 1] = 1.f - w0;
    atomicAdd(&counts[b0], 1);
    atomicAdd(&counts[b1], 1);
  }
}

__global__ void moe_init_kernel(int* counts, int* cursor, int* tok_of_slot, int cap) {
  const int i = blockIdx.x * 256 + threadIdx.x;
  if (i < 8) counts[i] = 0;
  else if (i < 16) cursor[i - 8] = 0;
  if (i < cap) tok_of_slot[i] = -1;
}

__global__ void moe_offsets_kernel(const int* counts, int* offsets, int* total) {
  if (threadIdx.x == 0 && blockIdx.x == 0) {
    int off = 0;
    for (int e = 0; e < 8; ++e) { offsets[e] = off; off += (counts[e] + 127) & ~127; }
    total[0] = off;
  }
}

__global__ void moe_assign_kernel(const int* __restrict__ ids, int* cursor,
                                  const int* __restrict__ offsets,
                                  int* __restrict__ tok_of_slot,
                                  int* __restrict__ slot_of_tok) {
  const int t = blockIdx.x * 256 + threadIdx.x;
  if (t >= T) return;
#pragma unroll
  for (int k2 = 0; k2 < 2; ++k2) {
    const int e = ids[t * 2 + k2];
    const int pos = atomicAdd(&cursor[e], 1);
    const int slot = offsets[e] + pos;
    tok_of_slot[slot] = t;
    slot_of_tok[t * 2 + k2] = slot;
  }
}

__global__ void moe_act_kernel(const float* __restrict__ h12,
                               unsigned short* __restrict__ act,
                               const int* __restrict__ total) {
  const int idx = blockIdx.x * 256 + threadIdx.x;
  const int slot = idx / FF, f = idx % FF;
  if (slot >= total[0]) return;
  const float g = h12[(long)slot * (2 * FF) + f];
  const float u = h12[(long)slot * (2 * FF) + FF + f];
  const float sig = 1.f / (1.f + __expf(-g));
  act[(long)slot * FF + f] = f2bf(g * sig * u);
}

__global__ void combine_kernel(const float* __restrict__ hres, const float* __restrict__ eo,
                               const int* __restrict__ slot_of_tok,
                               const float* __restrict__ wts, float* __restrict__ out) {
  const int t = blockIdx.x;
  const int s0 = slot_of_tok[t * 2], s1 = slot_of_tok[t * 2 + 1];
  const float w0 = wts[t * 2], w1 = wts[t * 2 + 1];
  for (int d = threadIdx.x; d < H; d += 256)
    out[(long)t * H + d] = hres[(long)t * H + d] + w0 * eo[(long)s0 * H + d] + w1 * eo[(long)s1 * H + d];
}

// ---------------- launch ----------------
extern "C" void kernel_launch(void* const* d_in, const int* in_sizes, int n_in,
                              void* d_out, int out_size, void* d_ws, size_t ws_size,
                              hipStream_t stream) {
  const float* hidden = (const float*)d_in[1];
  const float* rms1w  = (const float*)d_in[2];
  const float* rms2w  = (const float*)d_in[3];
  const float* wq     = (const float*)d_in[4];
  const float* wk     = (const float*)d_in[5];
  const float* wv     = (const float*)d_in[6];
  const float* wo     = (const float*)d_in[7];
  const float* sink   = (const float*)d_in[8];
  const float* gatew  = (const float*)d_in[9];
  const float* wgate  = (const float*)d_in[10];
  const float* wup    = (const float*)d_in[11];
  const float* wdown  = (const float*)d_in[12];
  float* out = (float*)d_out;
  char* ws = (char*)d_ws;
  auto U16 = [&](size_t o) { return (unsigned short*)(ws + o); };
  auto F32 = [&](size_t o) { return (float*)(ws + o); };
  auto I32 = [&](size_t o) { return (int*)(ws + o); };

  const dim3 tb(32, 8);
  // weight transpose+convert -> bf16 Bt layouts
  transpose_cvt_kernel<<<dim3(H/32, H/32), tb, 0, stream>>>(wq, U16(O_WQKVT), H, H, 0, 0);
  transpose_cvt_kernel<<<dim3((NKV*HD)/32, H/32), tb, 0, stream>>>(
      wk, U16(O_WQKVT) + (size_t)(NQ*HD)*H, H, NKV*HD, 0, 0);
  transpose_cvt_kernel<<<dim3((NKV*VD)/32, H/32), tb, 0, stream>>>(
      wv, U16(O_WQKVT) + (size_t)(NQ*HD + NKV*HD)*H, H, NKV*VD, 0, 0);
  transpose_cvt_kernel<<<dim3(H/32, (NQ*VD)/32), tb, 0, stream>>>(wo, U16(O_WOT), NQ*VD, H, 0, 0);
  transpose_cvt_kernel<<<dim3(FF/32, H/32, E), tb, 0, stream>>>(
      wgate, U16(O_WGUP), H, FF, (long)H*FF, (long)2*FF*H);
  transpose_cvt_kernel<<<dim3(FF/32, H/32, E), tb, 0, stream>>>(
      wup, U16(O_WGUP) + (size_t)FF*H, H, FF, (long)H*FF, (long)2*FF*H);
  transpose_cvt_kernel<<<dim3(H/32, FF/32, E), tb, 0, stream>>>(
      wdown, U16(O_WDN), FF, H, (long)FF*H, (long)H*FF);

  // attention path
  rmsnorm_kernel<<<T, 256, 0, stream>>>(hidden, rms1w, U16(O_XBF));
  gemm_bf16_kernel<false, false, false><<<dim3(NQKV/128, T/128), 256, 0, stream>>>(
      U16(O_XBF), U16(O_WQKVT), F32(O_QKV), NQKV, nullptr, H, 0, nullptr, nullptr, nullptr);
  rope_pack_kernel<<<T, 256, 0, stream>>>(F32(O_QKV), U16(O_QBF), U16(O_KBF), U16(O_VTBF));
  attn_kernel<<<dim3(T/16, NQ), 32, 0, stream>>>(
      U16(O_QBF), U16(O_KBF), U16(O_VTBF), sink, U16(O_ATTN));
  gemm_bf16_kernel<false, false, true><<<dim3(H/128, T/128), 256, 0, stream>>>(
      U16(O_ATTN), U16(O_WOT), F32(O_H), H, hidden, NQ*VD, 0, nullptr, nullptr, nullptr);

  // MoE path
  rmsnorm_kernel<<<T, 256, 0, stream>>>(F32(O_H), rms2w, U16(O_X2BF));
  moe_init_kernel<<<(CAP + 255) / 256, 256, 0, stream>>>(
      I32(O_COUNTS), I32(O_CURSOR), I32(O_TOKSLOT), CAP);
  router_kernel<<<T, 256, 0, stream>>>(U16(O_X2BF), gatew, I32(O_IDS), F32(O_WTS), I32(O_COUNTS));
  moe_offsets_kernel<<<1, 32, 0, stream>>>(I32(O_COUNTS), I32(O_OFFS), I32(O_TOTAL));
  moe_assign_kernel<<<(T + 255) / 256, 256, 0, stream>>>(
      I32(O_IDS), I32(O_CURSOR), I32(O_OFFS), I32(O_TOKSLOT), I32(O_SLOTTOK));
  gemm_bf16_kernel<true, true, false><<<dim3((2*FF)/128, T/128, E), 256, 0, stream>>>(
      U16(O_X2BF), U16(O_WGUP), F32(O_H12), 2*FF, nullptr, H, (long)2*FF*H,
      I32(O_COUNTS), I32(O_OFFS), I32(O_TOKSLOT));
  moe_act_kernel<<<(CAP * FF) / 256, 256, 0, stream>>>(F32(O_H12), U16(O_ACT), I32(O_TOTAL));
  gemm_bf16_kernel<true, false, false><<<dim3(H/128, T/128, E), 256, 0, stream>>>(
      U16(O_ACT), U16(O_WDN), F32(O_EO), H, nullptr, FF, (long)H*FF,
      I32(O_COUNTS), I32(O_OFFS), nullptr);
  combine_kernel<<<T, 256, 0, stream>>>(F32(O_H), F32(O_EO), I32(O_SLOTTOK), F32(O_WTS), out);
}
